// ELA_26061861552485
// MI455X (gfx1250) — matrix operand order repaired and checked by reading, not compiled, not run on hardware
//
#include <hip/hip_runtime.h>

typedef __bf16 bf16;
typedef __attribute__((ext_vector_type(16))) __bf16 v16bf;
typedef __attribute__((ext_vector_type(8)))  __bf16 v8bf;
typedef __attribute__((ext_vector_type(8)))  float  v8f;

#define EPSF 1e-6f

// ---------- helpers ----------
__device__ __forceinline__ float silu_f(float x) { return x / (1.f + __expf(-x)); }
__device__ __forceinline__ float sigm_f(float x) { return 1.f / (1.f + __expf(-x)); }

// A-matrix fragment (16x32 bf16), row-major source [row][k], stride in halves.
// ISA layout: lane holds row M=lane%16; halves 0..7 = K (8*hi)..(8*hi+7),
// halves 8..15 = K (16+8*hi)..(16+8*hi+7).
__device__ __forceinline__ v16bf frag_a(const bf16* base, int stride, int row, int k0) {
    int lane = threadIdx.x & 31;
    int hi   = lane >> 4;
    const bf16* p0 = base + (size_t)(row + (lane & 15)) * stride + k0 + 8 * hi;
    v8bf c0 = *(const v8bf*)p0;
    v8bf c1 = *(const v8bf*)(p0 + 16);
    v16bf a;
#pragma unroll
    for (int i = 0; i < 8; ++i) { a[i] = c0[i]; a[i + 8] = c1[i]; }
    return a;
}

// B-matrix fragment (32x16 bf16), source stored column-major-as-rows: [n][k].
// ISA layout: lane holds column N=lane%16; halves 0..7 = K (8*hi)..(8*hi+7),
// halves 8..15 = K (16+8*hi)..(16+8*hi+7).
__device__ __forceinline__ v16bf frag_b(const bf16* base, int stride, int col, int k0) {
    int lane = threadIdx.x & 31;
    int hi   = lane >> 4;
    const bf16* p0 = base + (size_t)(col + (lane & 15)) * stride + k0 + 8 * hi;
    v8bf c0 = *(const v8bf*)p0;
    v8bf c1 = *(const v8bf*)(p0 + 16);
    v16bf b;
#pragma unroll
    for (int i = 0; i < 8; ++i) { b[i] = c0[i]; b[i + 8] = c1[i]; }
    return b;
}

__device__ __forceinline__ v8f wmma_bf(v16bf a, v16bf b, v8f c) {
    return __builtin_amdgcn_wmma_f32_16x16x32_bf16(false, a, false, b, (short)0, c,
                                                   false, false);
}

// ---------- kernel 0: fp32 -> bf16 weight convert ----------
__global__ __launch_bounds__(256) void k_cvt(const float* __restrict__ s,
                                             bf16* __restrict__ d, int n) {
    int i = blockIdx.x * 256 + threadIdx.x;
    if (i < n) d[i] = (bf16)s[i];
}

// ---------- kernel 1: proj GEMM  y[b,o,n] = proj_w[o,:].x[b,:,n] + proj_b ----------
// grid: (49 nblk, 6 mblk, 16 b), block 256 (8 waves, 4x2 -> wave = 32x32)
__global__ __launch_bounds__(256) void k_proj(const float* __restrict__ x,
                                              const bf16* __restrict__ pw,
                                              const float* __restrict__ pb,
                                              float* __restrict__ y) {
    __shared__ bf16 Bt[64 * 200];   // [n][c] transposed x tile
    int b = blockIdx.z, mblk = blockIdx.y, nblk = blockIdx.x;
    int n0 = nblk * 64, m0b = mblk * 128;
    int tid = threadIdx.x;
    for (int i = tid; i < 192 * 64; i += 256) {
        int c = i >> 6, n = i & 63;
        Bt[n * 200 + c] = (bf16)x[((size_t)(b * 192 + c)) * 3136 + n0 + n];
    }
    __syncthreads();
    int w = tid >> 5, lane = tid & 31, hi = lane >> 4, ln = lane & 15;
    int m0  = m0b + (w >> 1) * 32;
    int nw0 = (w & 1) * 32;
    v8f acc[2][2] = {};
    for (int k = 0; k < 192; k += 32) {
        v16bf a0 = frag_a(pw, 192, m0, k);
        v16bf a1 = frag_a(pw, 192, m0 + 16, k);
        v16bf b0 = frag_b(Bt, 200, nw0, k);
        v16bf b1 = frag_b(Bt, 200, nw0 + 16, k);
        acc[0][0] = wmma_bf(a0, b0, acc[0][0]);
        acc[0][1] = wmma_bf(a0, b1, acc[0][1]);
        acc[1][0] = wmma_bf(a1, b0, acc[1][0]);
        acc[1][1] = wmma_bf(a1, b1, acc[1][1]);
    }
#pragma unroll
    for (int mi = 0; mi < 2; ++mi)
#pragma unroll
        for (int ni = 0; ni < 2; ++ni)
#pragma unroll
            for (int r = 0; r < 8; ++r) {
                int row = m0 + mi * 16 + r + 8 * hi;
                int col = n0 + nw0 + ni * 16 + ln;
                y[((size_t)(b * 768 + row)) * 3136 + col] = acc[mi][ni][r] + pb[row];
            }
}

// ---------- kernel 2: depthwise 3x3 + bias + residual ----------
__global__ __launch_bounds__(256) void k_dwc(const float* __restrict__ y,
                                             const float* __restrict__ dw,
                                             const float* __restrict__ db,
                                             float* __restrict__ xs) {
    size_t idx = (size_t)blockIdx.x * 256 + threadIdx.x;
    if (idx >= (size_t)16 * 768 * 3136) return;
    int wp = (int)(idx % 56);
    size_t t = idx / 56;
    int hp = (int)(t % 56);
    t /= 56;
    int ch = (int)(t % 768);
    int b  = (int)(t / 768);
    const float* wgt = dw + ch * 9;
    const float* yp  = y + ((size_t)(b * 768 + ch)) * 3136;
    float acc = 0.f;
#pragma unroll
    for (int i = 0; i < 3; ++i) {
        int hh = hp + i - 1;
        if (hh < 0 || hh >= 56) continue;
#pragma unroll
        for (int j = 0; j < 3; ++j) {
            int wq = wp + j - 1;
            if (wq < 0 || wq >= 56) continue;
            acc += wgt[i * 3 + j] * yp[hh * 56 + wq];
        }
    }
    xs[idx] = acc + db[ch] + y[idx];
}

// ---------- kernel 2b: kfmean[b,d] = mean_n silu(xs[b,192+d,n]) ----------
__global__ __launch_bounds__(256) void k_kfmean(const float* __restrict__ xs,
                                                float* __restrict__ kfm) {
    int d = blockIdx.x, b = blockIdx.y;
    const float* p = xs + ((size_t)(b * 768 + 192 + d)) * 3136;
    float acc = 0.f;
    for (int n = threadIdx.x; n < 3136; n += 256) acc += silu_f(p[n]);
    __shared__ float red[256];
    red[threadIdx.x] = acc;
    __syncthreads();
    for (int s = 128; s > 0; s >>= 1) {
        if ((int)threadIdx.x < s) red[threadIdx.x] += red[threadIdx.x + s];
        __syncthreads();
    }
    if (threadIdx.x == 0) kfm[b * 192 + d] = red[0] * (1.f / 3136.f);
}

// ---------- kernel 3: vk[b,v,d] = sum_n (v*s)(silu(k)*s) ----------
// grid: (3 dblk, 3 vblk, 16 b), block 256 (8 waves, 4x2 -> wave 32x32), K loop 3136/64
__global__ __launch_bounds__(256) void k_vk(const float* __restrict__ xs,
                                            bf16* __restrict__ vkb) {
    __shared__ bf16 At[128 * 72];   // [v][n]
    __shared__ bf16 Bt[64 * 72];    // [d][n]
    int b = blockIdx.z, mblk = blockIdx.y, nblk = blockIdx.x;
    int m0b = mblk * 128, d0 = nblk * 64;
    int tid = threadIdx.x, w = tid >> 5, lane = tid & 31, hi = lane >> 4, ln = lane & 15;
    int m0  = (w >> 1) * 32;
    int nw0 = (w & 1) * 32;
    const float s = 1.f / 56.f;
    v8f acc[2][2] = {};
    for (int n0 = 0; n0 < 3136; n0 += 64) {
        __syncthreads();
        for (int i = tid; i < 128 * 64; i += 256) {
            int m = i >> 6, kk = i & 63;
            At[m * 72 + kk] =
                (bf16)(xs[((size_t)(b * 768 + 384 + m0b + m)) * 3136 + n0 + kk] * s);
        }
        for (int i = tid; i < 64 * 64; i += 256) {
            int d = i >> 6, kk = i & 63;
            Bt[d * 72 + kk] =
                (bf16)(silu_f(xs[((size_t)(b * 768 + 192 + d0 + d)) * 3136 + n0 + kk]) * s);
        }
        __syncthreads();
#pragma unroll
        for (int k = 0; k < 64; k += 32) {
            v16bf a0 = frag_a(At, 72, m0, k);
            v16bf a1 = frag_a(At, 72, m0 + 16, k);
            v16bf b0 = frag_b(Bt, 72, nw0, k);
            v16bf b1 = frag_b(Bt, 72, nw0 + 16, k);
            acc[0][0] = wmma_bf(a0, b0, acc[0][0]);
            acc[0][1] = wmma_bf(a0, b1, acc[0][1]);
            acc[1][0] = wmma_bf(a1, b0, acc[1][0]);
            acc[1][1] = wmma_bf(a1, b1, acc[1][1]);
        }
    }
#pragma unroll
    for (int mi = 0; mi < 2; ++mi)
#pragma unroll
        for (int ni = 0; ni < 2; ++ni)
#pragma unroll
            for (int r = 0; r < 8; ++r) {
                int row = m0b + m0 + mi * 16 + r + 8 * hi;
                int dc  = d0 + nw0 + ni * 16 + ln;
                vkb[((size_t)(b * 384 + row)) * 192 + dc] = (bf16)acc[mi][ni][r];
            }
}

// ---------- kernel 4: fused attn/gate/rms/out, per (b, 32-column tile) ----------
// grid: (98 nblk, 16 b), block 256.  Dynamic LDS ~90KB (<< 320KB/WGP).
__global__ __launch_bounds__(256) void k_fused(const float* __restrict__ xs,
                                               const bf16* __restrict__ vkb,
                                               const float* __restrict__ kfm,
                                               const bf16* __restrict__ gwb,
                                               const float* __restrict__ gb,
                                               const float* __restrict__ anw,
                                               const float* __restrict__ svw,
                                               const bf16* __restrict__ owb,
                                               const float* __restrict__ outb,
                                               float* __restrict__ out) {
    extern __shared__ __align__(16) char smem_raw[];
    bf16* xs_t = (bf16*)smem_raw;            // [32][776]  all 768 channels, transposed
    bf16* qf_t = xs_t + 32 * 776;            // [32][200]  silu(q), transposed
    bf16* o_t  = qf_t + 32 * 200;            // [32][392]  gated attn -> final o
    float* z_s    = (float*)(o_t + 32 * 392);
    float* rinv_s = z_s + 32;
    float* rmsacc = rinv_s + 32;
    float* gb_s   = rmsacc + 32;             // [384]
    float* anw_s  = gb_s + 384;              // [384]
    float* sv_s   = anw_s + 384;             // [384]

    int b = blockIdx.y, n0 = blockIdx.x * 32;
    int tid = threadIdx.x;

    // ---- phase 0: stage ----
    for (int i = tid; i < 768 * 32; i += 256) {
        int c = i >> 5, n = i & 31;
        xs_t[n * 776 + c] = (bf16)xs[((size_t)(b * 768 + c)) * 3136 + n0 + n];
    }
    for (int i = tid; i < 192 * 32; i += 256) {
        int d = i >> 5, n = i & 31;
        qf_t[n * 200 + d] = (bf16)silu_f(xs[((size_t)(b * 768 + d)) * 3136 + n0 + n]);
    }
    for (int i = tid; i < 384; i += 256) {
        gb_s[i]  = gb[i];
        anw_s[i] = anw[i];
        sv_s[i]  = svw[i];
    }
    if (tid < 32) rmsacc[tid] = 0.f;
    __syncthreads();
    if (tid < 32) {
        const float* km = kfm + b * 192;
        float a = EPSF;
        for (int d = 0; d < 192; ++d) a += km[d] * (float)qf_t[tid * 200 + d];
        z_s[tid] = a;
    }
    __syncthreads();

    // ---- phase 1: attn & gate GEMMs (wave w -> rows [48w, 48w+48)) ----
    int w = tid >> 5, lane = tid & 31, hi = lane >> 4, ln = lane & 15;
    float ssq[2] = {0.f, 0.f};
    const bf16* vkb_b = vkb + (size_t)b * 384 * 192;
    for (int ms = 0; ms < 3; ++ms) {
        int m0 = w * 48 + ms * 16;
        v8f accA[2] = {};
        for (int k = 0; k < 192; k += 32) {
            v16bf a = frag_a(vkb_b, 192, m0, k);
            accA[0] = wmma_bf(a, frag_b(qf_t, 200, 0, k), accA[0]);
            accA[1] = wmma_bf(a, frag_b(qf_t, 200, 16, k), accA[1]);
        }
        v8f accG[2] = {};
        for (int k = 0; k < 768; k += 32) {
            v16bf a = frag_a(gwb, 768, m0, k);
            accG[0] = wmma_bf(a, frag_b(xs_t, 776, 0, k), accG[0]);
            accG[1] = wmma_bf(a, frag_b(xs_t, 776, 16, k), accG[1]);
        }
#pragma unroll
        for (int ni = 0; ni < 2; ++ni) {
            float zz = z_s[ni * 16 + ln];
#pragma unroll
            for (int r = 0; r < 8; ++r) {
                int row = m0 + r + 8 * hi;
                float g  = sigm_f(accG[ni][r] + gb_s[row]);
                float vv = (accA[ni][r] / zz) * g;
                ssq[ni] += vv * vv;
                o_t[(size_t)(ni * 16 + ln) * 392 + row] = (bf16)vv;
            }
        }
    }
    atomicAdd(&rmsacc[ln], ssq[0]);
    atomicAdd(&rmsacc[16 + ln], ssq[1]);
    __syncthreads();
    if (tid < 32) rinv_s[tid] = rsqrtf(rmsacc[tid] * (1.f / 384.f) + EPSF);
    __syncthreads();
    // ---- fixup: o = attn/rms*an_w + v*sv ----
    for (int i = tid; i < 384 * 32; i += 256) {
        int m = i >> 5, n = i & 31;
        float val = (float)o_t[n * 392 + m] * rinv_s[n] * anw_s[m] +
                    (float)xs_t[n * 776 + 384 + m] * sv_s[m];
        o_t[n * 392 + m] = (bf16)val;
    }
    __syncthreads();

    // ---- phase 2: out GEMM (192x384 @ 384x32), waves 4x2 ----
    int mg = w >> 1, ng = w & 1;
    v8f accO[3] = {};
    for (int k = 0; k < 384; k += 32) {
        v16bf bb = frag_b(o_t, 392, ng * 16, k);
#pragma unroll
        for (int mi = 0; mi < 3; ++mi) {
            v16bf a = frag_a(owb, 384, mg * 48 + mi * 16, k);
            accO[mi] = wmma_bf(a, bb, accO[mi]);
        }
    }
#pragma unroll
    for (int mi = 0; mi < 3; ++mi)
#pragma unroll
        for (int r = 0; r < 8; ++r) {
            int row = mg * 48 + mi * 16 + r + 8 * hi;
            int col = n0 + ng * 16 + ln;
            out[((size_t)(b * 192 + row)) * 3136 + col] = accO[mi][r] + outb[row];
        }
}

// ---------- host ----------
extern "C" void kernel_launch(void* const* d_in, const int* in_sizes, int n_in,
                              void* d_out, int out_size, void* d_ws, size_t ws_size,
                              hipStream_t stream) {
    const float* x      = (const float*)d_in[0];
    const float* proj_w = (const float*)d_in[1];
    const float* proj_b = (const float*)d_in[2];
    const float* dwc_w  = (const float*)d_in[3];
    const float* dwc_b  = (const float*)d_in[4];
    const float* g_w    = (const float*)d_in[5];
    const float* g_b    = (const float*)d_in[6];
    const float* an_w   = (const float*)d_in[7];
    const float* sv     = (const float*)d_in[8];
    const float* out_w  = (const float*)d_in[9];
    const float* out_b  = (const float*)d_in[10];
    float* out = (float*)d_out;

    char*  ws  = (char*)d_ws;
    size_t off = 0;
    auto take = [&](size_t bytes) -> void* {
        void* p = ws + off;
        off = (off + bytes + 255) & ~(size_t)255;
        return p;
    };
    const size_t YB = (size_t)16 * 768 * 3136 * 4;
    float* y   = (float*)take(YB);
    float* xs  = (float*)take(YB);
    bf16*  pwb = (bf16*)take((size_t)768 * 192 * 2);
    bf16*  gwb = (bf16*)take((size_t)384 * 768 * 2);
    bf16*  owb = (bf16*)take((size_t)192 * 384 * 2);
    bf16*  vkb = (bf16*)take((size_t)16 * 384 * 192 * 2);
    float* kfm = (float*)take((size_t)16 * 192 * 4);

    constexpr int SMEM_FUSED =
        (32 * 776 + 32 * 200 + 32 * 392) * 2 + (32 * 3 + 384 * 3) * 4;  // 92544 B
    hipFuncSetAttribute(reinterpret_cast<const void*>(k_fused),
                        hipFuncAttributeMaxDynamicSharedMemorySize, SMEM_FUSED);

    k_cvt<<<(768 * 192 + 255) / 256, 256, 0, stream>>>(proj_w, pwb, 768 * 192);
    k_cvt<<<(384 * 768 + 255) / 256, 256, 0, stream>>>(g_w, gwb, 384 * 768);
    k_cvt<<<(192 * 384 + 255) / 256, 256, 0, stream>>>(out_w, owb, 192 * 384);

    k_proj<<<dim3(49, 6, 16), 256, 0, stream>>>(x, pwb, proj_b, y);

    k_dwc<<<(int)(((size_t)16 * 768 * 3136 + 255) / 256), 256, 0, stream>>>(y, dwc_w,
                                                                            dwc_b, xs);

    k_kfmean<<<dim3(192, 16), 256, 0, stream>>>(xs, kfm);

    k_vk<<<dim3(3, 3, 16), 256, 0, stream>>>(xs, vkb);

    k_fused<<<dim3(98, 16), 256, SMEM_FUSED, stream>>>(xs, vkb, kfm, gwb, g_b, an_w, sv,
                                                       owb, out_b, out);
}
